// Mamba2Block_20942260535846
// MI455X (gfx1250) — matrix operand order repaired and checked by reading, not compiled, not run on hardware
//
#include <hip/hip_runtime.h>
#include <hip/hip_bf16.h>

// ---------------------------------------------------------------------------
// Problem constants (from the reference)
// ---------------------------------------------------------------------------
#define D_MODEL   1024
#define D_STATE   128
#define D_CONV    4
#define HEADDIM   64
#define D_INNER   2048
#define NHEADS    32
#define CONV_DIM  2304        // D_INNER + 2*D_STATE
#define D_IN_PROJ 4384        // 2*D_INNER + 2*D_STATE + NHEADS
#define BATCH     4
#define SEQLEN    2048
#define MTOT      (BATCH * SEQLEN)   // 8192 tokens
#define EPS_F     1e-5f

// ---------------------------------------------------------------------------
// WMMA types (gfx1250, wave32)
// ---------------------------------------------------------------------------
typedef __attribute__((ext_vector_type(16))) __bf16        v16bf;
typedef __attribute__((ext_vector_type(8)))  float         v8f;
typedef __attribute__((ext_vector_type(4)))  unsigned int  v4u;

__device__ __forceinline__ unsigned short f2bf(float f) {
    union { float f; unsigned int u; } x; x.f = f;
    unsigned int u = x.u;
    unsigned int r = u + 0x7FFFu + ((u >> 16) & 1u);   // round-to-nearest-even
    return (unsigned short)(r >> 16);
}

// ---------------------------------------------------------------------------
// bf16 WMMA GEMM:  C[M,N] = A[M,K] @ B[N,K]^T  (+ optional residual)
//
// block = 256 threads = 8 waves; wave w -> rows [blk.x*128 + w*16, +16),
// NSUB column subtiles of 16 per wave.  Branch-free K loop (ragged N handled
// by separate NSUB=4 / NSUB=2 launches) with one-deep software pipelining:
// fragments for iteration k+1 are loaded before the WMMAs of iteration k, so
// the tensor ops overlap the outstanding loads instead of draining loadcnt
// to zero each step.
//
// Fragment layouts (ISA 7.12.2, wave32):
//  A 16x32: lane L -> row L&15; halves 0-7 = K[(L>>4)*8..+7], halves 8-15 same+16
//  B 32x16: lane L -> col L&15; halves 0-7 = K[(L>>4)*8..+7], halves 8-15 same+16
//  C/D    : VGPR j, lane L -> M=(L>>4)*8+j, N=L&15
// ---------------------------------------------------------------------------
template<int NSUB, bool HAS_RESID>
__global__ void wmma_gemm_bf16(const unsigned short* __restrict__ A,
                               const unsigned short* __restrict__ B,
                               float* __restrict__ C,
                               const float* __restrict__ resid,
                               int N, int K, int n_base) {
    const int wave = threadIdx.x >> 5;
    const int lane = threadIdx.x & 31;
    const int m0 = (blockIdx.x * 8 + wave) * 16;
    const int n0 = n_base + blockIdx.y * (NSUB * 16);

    // Hoist per-lane fragment base pointers out of the K loop.
    const unsigned short* aptr =
        A + (size_t)(m0 + (lane & 15)) * K + (lane >> 4) * 8;
    const unsigned short* bptr[NSUB];
#pragma unroll
    for (int s = 0; s < NSUB; ++s)
        bptr[s] = B + (size_t)(n0 + s * 16 + (lane & 15)) * K + (lane >> 4) * 8;

    union Frag { v16bf bf; v4u u[2]; };

    // Prologue: fragments for k0 = 0.
    Frag a_cur;
    a_cur.u[0] = *(const v4u*)(aptr);
    a_cur.u[1] = *(const v4u*)(aptr + 16);
    Frag b_cur[NSUB];
#pragma unroll
    for (int s = 0; s < NSUB; ++s) {
        b_cur[s].u[0] = *(const v4u*)(bptr[s]);
        b_cur[s].u[1] = *(const v4u*)(bptr[s] + 16);
    }

    v8f acc[NSUB] = {};
#pragma unroll 2
    for (int k0 = 0; k0 < K - 32; k0 += 32) {
        // Issue next iteration's loads first...
        Frag a_nxt;
        a_nxt.u[0] = *(const v4u*)(aptr + k0 + 32);
        a_nxt.u[1] = *(const v4u*)(aptr + k0 + 48);
        __builtin_prefetch(aptr + k0 + 96, 0, 1);   // speculative, branch-free
        Frag b_nxt[NSUB];
#pragma unroll
        for (int s = 0; s < NSUB; ++s) {
            b_nxt[s].u[0] = *(const v4u*)(bptr[s] + k0 + 32);
            b_nxt[s].u[1] = *(const v4u*)(bptr[s] + k0 + 48);
        }
        // ...then consume the current fragments.
#pragma unroll
        for (int s = 0; s < NSUB; ++s)
            acc[s] = __builtin_amdgcn_wmma_f32_16x16x32_bf16(
                /*neg_a=*/false, a_cur.bf, /*neg_b=*/false, b_cur[s].bf,
                /*c_mod=*/(short)0, acc[s],
                /*reuse_a=*/false, /*reuse_b=*/false);
        a_cur = a_nxt;
#pragma unroll
        for (int s = 0; s < NSUB; ++s) b_cur[s] = b_nxt[s];
    }
    // Epilogue: last K step.
#pragma unroll
    for (int s = 0; s < NSUB; ++s)
        acc[s] = __builtin_amdgcn_wmma_f32_16x16x32_bf16(
            false, a_cur.bf, false, b_cur[s].bf, (short)0, acc[s], false, false);

    const int ncol   = lane & 15;
    const int rowoff = (lane >> 4) * 8;
#pragma unroll
    for (int s = 0; s < NSUB; ++s) {
        const int n = n0 + s * 16 + ncol;
#pragma unroll
        for (int j = 0; j < 8; ++j) {
            const int m = m0 + rowoff + j;
            float v = acc[s][j];
            if (HAS_RESID) v += resid[(size_t)m * N + n];
            C[(size_t)m * N + n] = v;
        }
    }
}

// ---------------------------------------------------------------------------
// fp32 -> bf16 cast (weights)
// ---------------------------------------------------------------------------
__global__ void cast_bf16_kernel(const float* __restrict__ in,
                                 unsigned short* __restrict__ out, long n) {
    long i = (long)blockIdx.x * blockDim.x + threadIdx.x;
    if (i < n) out[i] = f2bf(in[i]);
}

// ---------------------------------------------------------------------------
// LayerNorm over D_MODEL, emit bf16 activations for GEMM1. One block / token.
// ---------------------------------------------------------------------------
__global__ void layernorm_bf16_kernel(const float* __restrict__ x,
                                      const float* __restrict__ w,
                                      const float* __restrict__ b,
                                      unsigned short* __restrict__ u) {
    const int m = blockIdx.x, tid = threadIdx.x;
    __shared__ float r1[256], r2[256];
    const float* row = x + (size_t)m * D_MODEL;
    float s = 0.f, s2 = 0.f;
#pragma unroll
    for (int k = 0; k < D_MODEL / 256; ++k) {
        float v = row[tid + k * 256];
        s += v; s2 += v * v;
    }
    r1[tid] = s; r2[tid] = s2; __syncthreads();
    for (int st = 128; st > 0; st >>= 1) {
        if (tid < st) { r1[tid] += r1[tid + st]; r2[tid] += r2[tid + st]; }
        __syncthreads();
    }
    float mu  = r1[0] * (1.f / D_MODEL);
    float var = r2[0] * (1.f / D_MODEL) - mu * mu;
    float inv = rsqrtf(var + EPS_F);
#pragma unroll
    for (int k = 0; k < D_MODEL / 256; ++k) {
        int c = tid + k * 256;
        u[(size_t)m * D_MODEL + c] = f2bf((row[c] - mu) * inv * w[c] + b[c]);
    }
}

// ---------------------------------------------------------------------------
// Depthwise causal conv (4 taps) + SiLU over xBC columns of zxbcdt.
// One thread per (token, channel); consecutive tid -> consecutive channel.
// ---------------------------------------------------------------------------
__global__ void conv_silu_kernel(const float* __restrict__ zx,
                                 const float* __restrict__ cw,
                                 const float* __restrict__ cb,
                                 float* __restrict__ xconv) {
    long i = (long)blockIdx.x * blockDim.x + threadIdx.x;
    if (i >= (long)MTOT * CONV_DIM) return;
    int  c = (int)(i % CONV_DIM);
    long m = i / CONV_DIM;
    int  t = (int)(m % SEQLEN);
    long b = m / SEQLEN;
    float acc = cb[c];
#pragma unroll
    for (int j = 0; j < D_CONV; ++j) {
        int tt = t - (D_CONV - 1) + j;
        if (tt >= 0)
            acc += cw[c * D_CONV + j] *
                   zx[(size_t)(b * SEQLEN + tt) * D_IN_PROJ + D_INNER + c];
    }
    xconv[(size_t)m * CONV_DIM + c] = acc / (1.f + expf(-acc));  // SiLU
}

// ---------------------------------------------------------------------------
// dt = softplus(raw + dt_bias); dA = exp(dt * (-exp(A_log)))
// ---------------------------------------------------------------------------
__global__ void dt_kernel(const float* __restrict__ zx,
                          const float* __restrict__ dt_bias,
                          const float* __restrict__ A_log,
                          float* __restrict__ dtb, float* __restrict__ dab) {
    int i = blockIdx.x * blockDim.x + threadIdx.x;
    if (i >= MTOT * NHEADS) return;
    int h = i & (NHEADS - 1);
    int m = i >> 5;
    float raw = zx[(size_t)m * D_IN_PROJ + (D_IN_PROJ - NHEADS) + h] + dt_bias[h];
    float dt  = (raw > 20.f) ? raw : log1pf(expf(raw));
    float A   = -expf(A_log[h]);
    dtb[i] = dt;
    dab[i] = expf(dt * A);
}

// ---------------------------------------------------------------------------
// Selective state scan. One block per (batch, head); full 64x128 state lives
// in registers (32 f32 per lane, 256 lanes). Lane: p = tid&63, n-chunk = tid>>6.
// ---------------------------------------------------------------------------
__global__ void scan_kernel(const float* __restrict__ xconv,
                            const float* __restrict__ dtb,
                            const float* __restrict__ dab,
                            const float* __restrict__ Dp,
                            float* __restrict__ ywork) {
    const int bh = blockIdx.x;
    const int b  = bh >> 5;
    const int h  = bh & (NHEADS - 1);
    const int tid = threadIdx.x;
    const int p   = tid & (HEADDIM - 1);
    const int n0  = (tid >> 6) * 32;

    __shared__ float ldsB[D_STATE], ldsC[D_STATE], ldsX[HEADDIM], ldsP[256];
    float st[32];
#pragma unroll
    for (int j = 0; j < 32; ++j) st[j] = 0.f;
    const float Dh = Dp[h];

    for (int t = 0; t < SEQLEN; ++t) {
        const size_t m = (size_t)b * SEQLEN + t;
        const float* row = xconv + m * CONV_DIM;
        if (tid < D_STATE) ldsB[tid] = row[D_INNER + tid];
        else               ldsC[tid - D_STATE] = row[D_INNER + D_STATE + (tid - D_STATE)];
        if (tid < HEADDIM) ldsX[tid] = row[h * HEADDIM + tid];
        const float s_dA = dab[m * NHEADS + h];
        const float s_dt = dtb[m * NHEADS + h];
        __syncthreads();

        const float coeff = s_dt * ldsX[p];
        float acc = 0.f;
#pragma unroll
        for (int j = 0; j < 32; ++j) {
            st[j] = st[j] * s_dA + coeff * ldsB[n0 + j];
            acc  += st[j] * ldsC[n0 + j];
        }
        ldsP[tid] = acc;
        __syncthreads();

        if (tid < HEADDIM) {
            float y = ldsP[tid] + ldsP[64 + tid] + ldsP[128 + tid] + ldsP[192 + tid]
                    + Dh * ldsX[tid];
            ywork[m * D_INNER + h * HEADDIM + tid] = y;
        }
        __syncthreads();   // protect LDS before next step's loads
    }
}

// ---------------------------------------------------------------------------
// y * silu(z), RMSNorm with gamma, emit bf16 for GEMM2. One block / token.
// ---------------------------------------------------------------------------
__global__ void gate_rms_kernel(const float* __restrict__ ywork,
                                const float* __restrict__ zx,
                                const float* __restrict__ gamma,
                                unsigned short* __restrict__ ybf) {
    const int m = blockIdx.x, tid = threadIdx.x;
    __shared__ float red[256];
    const float* yr = ywork + (size_t)m * D_INNER;
    const float* zr = zx + (size_t)m * D_IN_PROJ;   // z = first D_INNER cols
    float vals[D_INNER / 256];
    float ssq = 0.f;
#pragma unroll
    for (int k = 0; k < D_INNER / 256; ++k) {
        int c = tid + k * 256;
        float z = zr[c];
        float v = yr[c] * (z / (1.f + expf(-z)));
        vals[k] = v;
        ssq += v * v;
    }
    red[tid] = ssq; __syncthreads();
    for (int s = 128; s > 0; s >>= 1) {
        if (tid < s) red[tid] += red[tid + s];
        __syncthreads();
    }
    float inv = rsqrtf(red[0] * (1.f / D_INNER) + EPS_F);
#pragma unroll
    for (int k = 0; k < D_INNER / 256; ++k) {
        int c = tid + k * 256;
        ybf[(size_t)m * D_INNER + c] = f2bf(vals[k] * inv * gamma[c]);
    }
}

// ---------------------------------------------------------------------------
// Workspace layout (bytes, all 256B-aligned by construction)
// ---------------------------------------------------------------------------
static constexpr size_t SZ_U  = (size_t)MTOT * D_MODEL * 2;        // u bf16
static constexpr size_t SZ_W1 = (size_t)D_IN_PROJ * D_MODEL * 2;   // in_proj bf16
static constexpr size_t SZ_W2 = (size_t)D_MODEL * D_INNER * 2;     // out_proj bf16
static constexpr size_t SZ_ZX = (size_t)MTOT * D_IN_PROJ * 4;      // zxbcdt f32
static constexpr size_t SZ_XC = (size_t)MTOT * CONV_DIM * 4;       // conv+silu f32
static constexpr size_t SZ_DT = (size_t)MTOT * NHEADS * 4;         // dt f32
static constexpr size_t SZ_DA = (size_t)MTOT * NHEADS * 4;         // dA f32
static constexpr size_t SZ_Y  = (size_t)MTOT * D_INNER * 4;        // scan out f32
static constexpr size_t OFF_U  = 0;
static constexpr size_t OFF_W1 = OFF_U  + SZ_U;
static constexpr size_t OFF_W2 = OFF_W1 + SZ_W1;
static constexpr size_t OFF_ZX = OFF_W2 + SZ_W2;
static constexpr size_t OFF_XC = OFF_ZX + SZ_ZX;
static constexpr size_t OFF_DT = OFF_XC + SZ_XC;
static constexpr size_t OFF_DA = OFF_DT + SZ_DT;
static constexpr size_t OFF_Y  = OFF_DA + SZ_DA;
static constexpr size_t OFF_YB = OFF_Y  + SZ_Y;

extern "C" void kernel_launch(void* const* d_in, const int* in_sizes, int n_in,
                              void* d_out, int out_size, void* d_ws, size_t ws_size,
                              hipStream_t stream) {
    const float* x          = (const float*)d_in[0];
    const float* ln_w       = (const float*)d_in[1];
    const float* ln_b       = (const float*)d_in[2];
    const float* in_proj_w  = (const float*)d_in[3];
    const float* conv_w     = (const float*)d_in[4];
    const float* conv_b     = (const float*)d_in[5];
    const float* A_log      = (const float*)d_in[6];
    const float* Dp         = (const float*)d_in[7];
    const float* dt_bias    = (const float*)d_in[8];
    const float* gamma      = (const float*)d_in[9];
    const float* out_proj_w = (const float*)d_in[10];
    float* out = (float*)d_out;

    char* ws = (char*)d_ws;
    unsigned short* u_bf  = (unsigned short*)(ws + OFF_U);
    unsigned short* w1_bf = (unsigned short*)(ws + OFF_W1);
    unsigned short* w2_bf = (unsigned short*)(ws + OFF_W2);
    float*          zx    = (float*)(ws + OFF_ZX);
    float*          xconv = (float*)(ws + OFF_XC);
    float*          dtb   = (float*)(ws + OFF_DT);
    float*          dab   = (float*)(ws + OFF_DA);
    float*          ywork = (float*)(ws + OFF_Y);
    unsigned short* y_bf  = (unsigned short*)(ws + OFF_YB);

    // 1. weights -> bf16
    long n1 = (long)D_IN_PROJ * D_MODEL;
    cast_bf16_kernel<<<(int)((n1 + 255) / 256), 256, 0, stream>>>(in_proj_w, w1_bf, n1);
    long n2 = (long)D_MODEL * D_INNER;
    cast_bf16_kernel<<<(int)((n2 + 255) / 256), 256, 0, stream>>>(out_proj_w, w2_bf, n2);

    // 2. LayerNorm -> bf16 activations
    layernorm_bf16_kernel<<<MTOT, 256, 0, stream>>>(x, ln_w, ln_b, u_bf);

    // 3. in_proj GEMM: zxbcdt[8192,4384] = u @ W1^T  (WMMA bf16)
    //    main: 68 full 64-wide column blocks; tail: 2 subtiles for [4352,4384)
    dim3 g1a(MTOT / 128, (D_IN_PROJ - 32) / 64);   // 68 blocks -> cols [0,4352)
    wmma_gemm_bf16<4, false><<<g1a, 256, 0, stream>>>(u_bf, w1_bf, zx, nullptr,
                                                      D_IN_PROJ, D_MODEL, 0);
    dim3 g1b(MTOT / 128, 1);                       // cols [4352,4384)
    wmma_gemm_bf16<2, false><<<g1b, 256, 0, stream>>>(u_bf, w1_bf, zx, nullptr,
                                                      D_IN_PROJ, D_MODEL,
                                                      D_IN_PROJ - 32);

    // 4. depthwise conv + SiLU on xBC slice
    long nc = (long)MTOT * CONV_DIM;
    conv_silu_kernel<<<(int)((nc + 255) / 256), 256, 0, stream>>>(zx, conv_w, conv_b, xconv);

    // 5. dt / dA
    int nd = MTOT * NHEADS;
    dt_kernel<<<(nd + 255) / 256, 256, 0, stream>>>(zx, dt_bias, A_log, dtb, dab);

    // 6. sequential selective scan (state in registers)
    scan_kernel<<<BATCH * NHEADS, 256, 0, stream>>>(xconv, dtb, dab, Dp, ywork);

    // 7. gate + RMSNorm -> bf16
    gate_rms_kernel<<<MTOT, 256, 0, stream>>>(ywork, zx, gamma, y_bf);

    // 8. out_proj GEMM + residual: out = x + y @ W2^T
    dim3 g2(MTOT / 128, D_MODEL / 64);
    wmma_gemm_bf16<4, true><<<g2, 256, 0, stream>>>(y_bf, w2_bf, out, x,
                                                    D_MODEL, D_INNER, 0);
}
